// WMSA_13932873908942
// MI455X (gfx1250) — compile-verified, matrix operand order checked by reading, not run on hardware
//
#include <hip/hip_runtime.h>
#include <hip/hip_bf16.h>

// ---------------------------------------------------------------------------
// 3D shifted-window MSA (Swin) on gfx1250.
//   B=2, S=32^3, C=192, 6 heads x 32, WIN=4 (64 tok/window), SHIFT=2.
//   All GEMMs use v_wmma_f32_16x16x32_bf16 (HEAD_DIM==32 == WMMA K).
//   All WMMA B-operands are staged N-major in LDS so both A and B fragment
//   loads vectorize to ds_load_b128; A-fragments are hoisted into registers
//   and reused across all N-slabs.
// ---------------------------------------------------------------------------

typedef __attribute__((ext_vector_type(16))) __bf16 v16bf;
typedef __attribute__((ext_vector_type(8)))  float  v8f;

// k index inside a 16x32 bf16 A/B fragment for element e, lane-half hi.
__device__ __forceinline__ int frag_k(int e, int hi) {
    return (e < 8) ? (hi * 8 + e) : (8 + hi * 8 + e);
}

__device__ __forceinline__ unsigned pack_bf2(float a, float b) {
    union { __bf16 h[2]; unsigned u; } un;
    un.h[0] = (__bf16)a; un.h[1] = (__bf16)b;
    return un.u;
}

// ===========================================================================
// Kernel 1: fused roll + window-partition gather, QKV projection.
//   grid = B*512 (one window per block), 256 threads = 8 waves.
//   A (64x192) in LDS bf16, A-fragments hoisted to registers (pair-invariant).
//   W_qkv staged transposed (N-major) 32 cols at a time. The 32-col slab of
//   iteration `pair` maps to exactly one Q/K/V head group, so store routing
//   is a *uniform* branch on pair (no exec divergence).
// ===========================================================================
__global__ __launch_bounds__(256) void qkv_kernel(
    const float* __restrict__ x, const float* __restrict__ Wqkv,
    const float* __restrict__ bqkv,
    __bf16* __restrict__ Qb, __bf16* __restrict__ Kb, __bf16* __restrict__ Vb)
{
    __shared__ __bf16 As[64][200];   // [token][chan]   stride 400B (16B mult)
    __shared__ __bf16 Bs[32][200];   // [outcol][chan]  transposed weight slab

    const int tid = threadIdx.x;
    const int win = blockIdx.x & 511;
    const int b   = blockIdx.x >> 9;
    const int w1 = win >> 6, w2 = (win >> 3) & 7, w3 = win & 7;

    __builtin_prefetch(Wqkv, 0, 1);  // global_prefetch_b8 (L2 warm hint)

    // gather rolled+partitioned tile, fp32 -> bf16 (2 channels per store)
    for (int idx = tid; idx < 64 * 96; idx += 256) {
        int t = idx / 96, c = (idx - t * 96) * 2;
        int i = t >> 4, j = (t >> 2) & 3, k = t & 3;
        int z  = (w1 * 4 + i + 2) & 31;   // roll by -SHIFT folded into gather
        int y  = (w2 * 4 + j + 2) & 31;
        int xx = (w3 * 4 + k + 2) & 31;
        const float* src =
            &x[((((size_t)b * 32 + z) * 32 + y) * 32 + xx) * 192 + c];
        *(unsigned*)&As[t][c] = pack_bf2(src[0], src[1]);
    }
    __syncthreads();

    const int wid  = tid >> 5, lane = tid & 31;
    const int mt   = wid & 3;        // M tile (16 tokens)
    const int nsub = wid >> 2;       // which 16-col half of the 32-col slab
    const int mrow = lane & 15, hi = lane >> 4;
    const float scale = 0.17677669529663687f;  // 1/sqrt(32)

    // hoist all six A-fragments (invariant across the 18 N-slabs)
    v16bf afrag[6];
    #pragma unroll
    for (int kt = 0; kt < 6; ++kt)
        #pragma unroll
        for (int e = 0; e < 16; ++e)
            afrag[kt][e] = As[mt * 16 + mrow][kt * 32 + frag_k(e, hi)];

    const int c = nsub * 16 + mrow;          // output column within slab
    for (int pair = 0; pair < 18; ++pair) {  // 18 * 32 = 576 output cols
        __syncthreads();
        const int nbase = pair * 32;
        // stage W_qkv[:, nbase:nbase+32] transposed: Bs[n][k], 2 k per store
        for (int idx = tid; idx < 96 * 32; idx += 256) {
            int r = (idx >> 5) * 2, cc = idx & 31;    // coalesced global rows
            float v0 = Wqkv[(size_t)r * 576 + nbase + cc];
            float v1 = Wqkv[(size_t)(r + 1) * 576 + nbase + cc];
            *(unsigned*)&Bs[cc][r] = pack_bf2(v0, v1);
        }
        __syncthreads();

        v8f acc = {};
        #pragma unroll
        for (int kt = 0; kt < 6; ++kt) {     // K = 192 = 6 x 32
            v16bf bf;
            #pragma unroll
            for (int e = 0; e < 16; ++e)
                bf[e] = Bs[nsub * 16 + mrow][kt * 32 + frag_k(e, hi)];
            acc = __builtin_amdgcn_wmma_f32_16x16x32_bf16(
                false, afrag[kt], false, bf, (short)0, acc, false, false);
        }

        // uniform store routing: slab `pair` is head-group `pair`
        const float bias = bqkv[nbase + c];
        if (pair < 6) {
            __bf16* dst = Qb + ((size_t)blockIdx.x * 6 + pair) * 2048;
            #pragma unroll
            for (int e = 0; e < 8; ++e) {
                int t = mt * 16 + hi * 8 + e;
                dst[t * 32 + c] = (__bf16)((acc[e] + bias) * scale);
            }
        } else if (pair < 12) {
            __bf16* dst = Kb + ((size_t)blockIdx.x * 6 + (pair - 6)) * 2048;
            #pragma unroll
            for (int e = 0; e < 8; ++e) {
                int t = mt * 16 + hi * 8 + e;
                dst[t * 32 + c] = (__bf16)(acc[e] + bias);     // row-major
            }
        } else {
            __bf16* dst = Vb + ((size_t)blockIdx.x * 6 + (pair - 12)) * 2048;
            #pragma unroll
            for (int e = 0; e < 8; ++e) {
                int t = mt * 16 + hi * 8 + e;
                dst[c * 64 + t] = (__bf16)(acc[e] + bias);     // channel-major
            }
        }
    }
}

// ===========================================================================
// Kernel 2: windowed attention. grid = B*512*6 (window,head), 128 thr = 4 waves.
//   S = Q K^T: one WMMA per 16x16 tile (K == HEAD_DIM == 32).
//   bias from rel_params (LDS), analytic shift mask, softmax via shfl_xor.
//   P re-shaped through padded LDS into A-fragments for P @ V.
// ===========================================================================
__global__ __launch_bounds__(128) void attn_kernel(
    const __bf16* __restrict__ Qb, const __bf16* __restrict__ Kb,
    const __bf16* __restrict__ Vb, const float* __restrict__ rel,
    __bf16* __restrict__ Ob)
{
    __shared__ float  biasS[343];        // 7^3 rel-pos table for this head
    __shared__ __bf16 Ql[64][40];        // [token][chan]   stride 80B
    __shared__ __bf16 Kl[64][40];        // [token][chan]   (N-major for QK^T)
    __shared__ __bf16 Vt[32][72];        // [chan][token]   (N-major for PV)
    __shared__ __bf16 Ps[4][16][72];     // per-wave probs, stride 144B

    const int tid = threadIdx.x;
    const int h   = blockIdx.x % 6;
    const size_t gw = blockIdx.x / 6;          // global window (incl. batch)
    const int win = (int)(gw & 511);
    const int w1 = win >> 6, w2 = (win >> 3) & 7, w3 = win & 7;

    const uint4* qp = (const uint4*)(Qb + (size_t)blockIdx.x * 2048);
    const uint4* kp = (const uint4*)(Kb + (size_t)blockIdx.x * 2048);
    const uint4* vp = (const uint4*)(Vb + (size_t)blockIdx.x * 2048);

    for (int idx = tid; idx < 343; idx += 128)
        biasS[idx] = rel[(size_t)h * 343 + idx];
    // 2048 bf16 = 256 uint4 per array; LDS rows padded to 16B multiples
    for (int idx = tid; idx < 256; idx += 128) {
        int rq = idx >> 2, cq = (idx & 3) * 8;       // 4 uint4 per 32-ch row
        *(uint4*)&Ql[rq][cq] = qp[idx];
        *(uint4*)&Kl[rq][cq] = kp[idx];
        int rv = idx >> 3, cv = (idx & 7) * 8;       // 8 uint4 per 64-tok row
        *(uint4*)&Vt[rv][cv] = vp[idx];
    }
    __syncthreads();

    const int wid = tid >> 5, lane = tid & 31;
    const int mrow = lane & 15, hi = lane >> 4;
    const int t0 = wid * 16;                 // this wave's 16 query tokens

    // --- S = Q K^T ---------------------------------------------------------
    v16bf a;
    #pragma unroll
    for (int e = 0; e < 16; ++e)
        a[e] = Ql[t0 + mrow][frag_k(e, hi)];

    v8f accS[4];
    #pragma unroll
    for (int f = 0; f < 4; ++f) {
        v16bf bf;
        #pragma unroll
        for (int e = 0; e < 16; ++e)
            bf[e] = Kl[f * 16 + mrow][frag_k(e, hi)];   // contiguous in k
        v8f z = {};
        accS[f] = __builtin_amdgcn_wmma_f32_16x16x32_bf16(
            false, a, false, bf, (short)0, z, false, false);
    }

    // --- bias + shift mask -------------------------------------------------
    float sv[4][8];
    #pragma unroll
    for (int f = 0; f < 4; ++f) {
        int q = f * 16 + mrow;
        int i2 = q >> 4, j2 = (q >> 2) & 3, k2 = q & 3;
        #pragma unroll
        for (int e = 0; e < 8; ++e) {
            int p = t0 + hi * 8 + e;
            int i1 = p >> 4, j1 = (p >> 2) & 3, k1 = p & 3;
            float v = accS[f][e] +
                biasS[((i1 - i2 + 3) * 7 + (j1 - j2 + 3)) * 7 + (k1 - k2 + 3)];
            bool msk = (w1 == 7 && ((i1 < 2) != (i2 < 2))) ||
                       (w2 == 7 && ((j1 < 2) != (j2 < 2))) ||
                       (w3 == 7 && ((k1 < 2) != (k2 < 2)));
            sv[f][e] = msk ? -3.0e38f : v;
        }
    }

    // --- softmax: row lives in elem e across 16 lanes of one half ----------
    #pragma unroll
    for (int e = 0; e < 8; ++e) {
        float m = fmaxf(fmaxf(sv[0][e], sv[1][e]), fmaxf(sv[2][e], sv[3][e]));
        #pragma unroll
        for (int d = 1; d < 16; d <<= 1) m = fmaxf(m, __shfl_xor(m, d, 32));
        float s = 0.f;
        #pragma unroll
        for (int f = 0; f < 4; ++f) { sv[f][e] = __expf(sv[f][e] - m); s += sv[f][e]; }
        #pragma unroll
        for (int d = 1; d < 16; d <<= 1) s += __shfl_xor(s, d, 32);
        float inv = 1.0f / s;
        #pragma unroll
        for (int f = 0; f < 4; ++f) sv[f][e] *= inv;
    }

    // stash probs (C-layout) into LDS so they can be re-read as A-fragments
    #pragma unroll
    for (int f = 0; f < 4; ++f)
        #pragma unroll
        for (int e = 0; e < 8; ++e)
            Ps[wid][hi * 8 + e][f * 16 + mrow] = (__bf16)sv[f][e];
    __syncthreads();

    // --- O = P @ V ---------------------------------------------------------
    v8f accO[2] = {};
    #pragma unroll
    for (int kt = 0; kt < 2; ++kt) {        // K = 64 query keys = 2 x 32
        v16bf a2;
        #pragma unroll
        for (int e = 0; e < 16; ++e)
            a2[e] = Ps[wid][mrow][kt * 32 + frag_k(e, hi)];
        #pragma unroll
        for (int f = 0; f < 2; ++f) {
            v16bf bf;
            #pragma unroll
            for (int e = 0; e < 16; ++e)
                bf[e] = Vt[f * 16 + mrow][kt * 32 + frag_k(e, hi)];
            accO[f] = __builtin_amdgcn_wmma_f32_16x16x32_bf16(
                false, a2, false, bf, (short)0, accO[f], false, false);
        }
    }

    // store head-concatenated output rows (bf16)
    #pragma unroll
    for (int f = 0; f < 2; ++f) {
        int cc = h * 32 + f * 16 + mrow;
        #pragma unroll
        for (int e = 0; e < 8; ++e) {
            int t = t0 + hi * 8 + e;
            Ob[(gw * 64 + t) * 192 + cc] = (__bf16)accO[f][e];
        }
    }
}

// ===========================================================================
// Kernel 3: output projection + fused inverse-partition / roll scatter.
//   grid = B*512, 256 threads = 8 waves. A-fragments hoisted to registers.
// ===========================================================================
__global__ __launch_bounds__(256) void proj_kernel(
    const __bf16* __restrict__ Ob, const float* __restrict__ Wout,
    const float* __restrict__ bout, float* __restrict__ out)
{
    __shared__ __bf16 As[64][200];   // [token][chan]
    __shared__ __bf16 Bs[32][200];   // [outcol][chan] transposed weight slab

    const int tid = threadIdx.x;
    const int win = blockIdx.x & 511;
    const int b   = blockIdx.x >> 9;
    const int w1 = win >> 6, w2 = (win >> 3) & 7, w3 = win & 7;

    // A copy: Ob already bf16 row-major -> wide uint4 copy into padded LDS
    const uint4* op = (const uint4*)(Ob + (size_t)blockIdx.x * 64 * 192);
    for (int idx = tid; idx < 1536; idx += 256) {     // 64*192*2B / 16B
        int t = idx / 24, q = (idx - t * 24) * 8;     // 24 uint4 per row
        *(uint4*)&As[t][q] = op[idx];
    }
    __syncthreads();

    const int wid = tid >> 5, lane = tid & 31;
    const int mt = wid & 3, nsub = wid >> 2;
    const int mrow = lane & 15, hi = lane >> 4;

    v16bf afrag[6];
    #pragma unroll
    for (int kt = 0; kt < 6; ++kt)
        #pragma unroll
        for (int e = 0; e < 16; ++e)
            afrag[kt][e] = As[mt * 16 + mrow][kt * 32 + frag_k(e, hi)];

    const int c = nsub * 16 + mrow;
    for (int pair = 0; pair < 6; ++pair) {    // 6 * 32 = 192 output cols
        __syncthreads();
        const int nbase = pair * 32;
        for (int idx = tid; idx < 96 * 32; idx += 256) {
            int r = (idx >> 5) * 2, cc = idx & 31;
            float v0 = Wout[(size_t)r * 192 + nbase + cc];
            float v1 = Wout[(size_t)(r + 1) * 192 + nbase + cc];
            *(unsigned*)&Bs[cc][r] = pack_bf2(v0, v1);
        }
        __syncthreads();

        v8f acc = {};
        #pragma unroll
        for (int kt = 0; kt < 6; ++kt) {
            v16bf bf;
            #pragma unroll
            for (int e = 0; e < 16; ++e)
                bf[e] = Bs[nsub * 16 + mrow][kt * 32 + frag_k(e, hi)];
            acc = __builtin_amdgcn_wmma_f32_16x16x32_bf16(
                false, afrag[kt], false, bf, (short)0, acc, false, false);
        }

        const int jcol = nbase + c;
        const float bj = bout[jcol];
        #pragma unroll
        for (int e = 0; e < 8; ++e) {
            int t = mt * 16 + hi * 8 + e;
            int i = t >> 4, j = (t >> 2) & 3, k = t & 3;
            int z  = (w1 * 4 + i + 2) & 31;   // inverse partition + roll(+2)
            int y  = (w2 * 4 + j + 2) & 31;
            int xx = (w3 * 4 + k + 2) & 31;
            out[((((size_t)b * 32 + z) * 32 + y) * 32 + xx) * 192 + jcol] =
                acc[e] + bj;
        }
    }
}

// ===========================================================================
extern "C" void kernel_launch(void* const* d_in, const int* in_sizes, int n_in,
                              void* d_out, int out_size, void* d_ws, size_t ws_size,
                              hipStream_t stream)
{
    const float* x    = (const float*)d_in[0];
    const float* Wqkv = (const float*)d_in[1];
    const float* bqkv = (const float*)d_in[2];
    const float* rel  = (const float*)d_in[3];
    const float* Wout = (const float*)d_in[4];
    const float* bout = (const float*)d_in[5];
    float* out = (float*)d_out;

    const int NWIN = 2 * 512;                       // B * nw^3
    const size_t qkvElems = (size_t)NWIN * 6 * 64 * 32;  // per Q/K/V array

    __bf16* Qb = (__bf16*)d_ws;
    __bf16* Kb = Qb + qkvElems;
    __bf16* Vb = Kb + qkvElems;
    __bf16* Ob = Vb + qkvElems;                     // NWIN * 64 * 192

    qkv_kernel <<<NWIN,     256, 0, stream>>>(x, Wqkv, bqkv, Qb, Kb, Vb);
    attn_kernel<<<NWIN * 6, 128, 0, stream>>>(Qb, Kb, Vb, rel, Ob);
    proj_kernel<<<NWIN,     256, 0, stream>>>(Ob, Wout, bout, out);
}
